// Attention_6219112645023
// MI455X (gfx1250) — compile-verified, matrix operand order checked by reading, not run on hardware
//
#include <hip/hip_runtime.h>
#include <hip/hip_bf16.h>
#include <math.h>

#define DIMC 2048
#define NH   32
#define NKV  8
#define HD   64
#define BB   2
#define TT   2048
#define MROWS (BB*TT)        // 4096
#define KVDIM (NKV*HD)       // 512
#define NEGBIG -1e30f

// CDNA5 async global->LDS staging (ASYNCcnt path), with safe fallback.
#if defined(__gfx1250__) && __has_builtin(__builtin_amdgcn_global_load_async_to_lds_b128) && __has_builtin(__builtin_amdgcn_s_wait_asynccnt)
#define USE_ASYNC_LDS 1
#else
#define USE_ASYNC_LDS 0
#endif

typedef __attribute__((ext_vector_type(16))) __bf16 v16bf;
typedef __attribute__((ext_vector_type(8)))  float  v8f;
typedef __attribute__((ext_vector_type(4)))  int    v4i;

#if USE_ASYNC_LDS
typedef __attribute__((address_space(1))) v4i* as1_v4i_ptr;
typedef __attribute__((address_space(3))) v4i* as3_v4i_ptr;
#endif

union FragBF { v16bf v; uint4 u2[2]; };

__device__ __forceinline__ unsigned short f2bf(float x) {
    unsigned int u = __float_as_uint(x);
    unsigned int r = (u + 0x7FFFu + ((u >> 16) & 1u)) >> 16;
    return (unsigned short)r;
}

__device__ __forceinline__ v8f wmma_bf16(v16bf a, v16bf b, v8f c) {
    return __builtin_amdgcn_wmma_f32_16x16x32_bf16(false, a, false, b, (short)0, c, false, false);
}

// 16x32 bf16 fragment (A layout; B layout identical with rows = N) from a
// K-contiguous row-major matrix. ISA 7.12.2: lanes 0-15 hold K {0-7,16-23},
// lanes 16-31 hold K {8-15,24-31}, row/col = lane&15.
__device__ __forceinline__ v16bf load_frag_k8(const unsigned short* base, int stride,
                                              int row0, int k0, int lane) {
    int r  = lane & 15;
    int hi = (lane >> 4) & 1;
    const unsigned short* p = base + (size_t)(row0 + r) * stride + k0 + hi * 8;
    FragBF f;
    f.u2[0] = *reinterpret_cast<const uint4*>(p);
    f.u2[1] = *reinterpret_cast<const uint4*>(p + 16);
    return f.v;
}

// Same fragment layout, but from a [rows][32] LDS tile.
__device__ __forceinline__ v16bf load_frag_lds(const unsigned short* tile, int row0, int lane) {
    int r  = lane & 15;
    int hi = (lane >> 4) & 1;
    const unsigned short* p = tile + (row0 + r) * 32 + hi * 8;
    FragBF f;
    f.u2[0] = *reinterpret_cast<const uint4*>(p);
    f.u2[1] = *reinterpret_cast<const uint4*>(p + 16);
    return f.v;
}

// Stage a 64x32 bf16 B tile into LDS: 256 threads x 16B each.
__device__ __forceinline__ void stage_b_tile(const unsigned short* __restrict__ Bm,
                                             unsigned short* lds_tile,
                                             int n0, int kk, int K, int tid) {
    int row = tid >> 2;                 // 0..63
    int ch  = tid & 3;                  // 0..3, 8 elements each
    const unsigned short* g = Bm + (size_t)(n0 + row) * K + kk + ch * 8;
    unsigned short* l = lds_tile + row * 32 + ch * 8;
#if USE_ASYNC_LDS
    __builtin_amdgcn_global_load_async_to_lds_b128(
        (as1_v4i_ptr)g, (as3_v4i_ptr)l, 0, 0);
#else
    *reinterpret_cast<uint4*>(l) = *reinterpret_cast<const uint4*>(g);
#endif
}

__device__ __forceinline__ void wait_stage() {
#if USE_ASYNC_LDS
    __builtin_amdgcn_s_wait_asynccnt(0);
#endif
}

// ---------------- fp32 -> bf16 convert ----------------
__global__ void cvt_f32_bf16(const float* __restrict__ src, unsigned short* __restrict__ dst, int n) {
    int i = blockIdx.x * 256 + threadIdx.x;
    if (i < n) dst[i] = f2bf(src[i]);
}

// ---------------- bf16 NT GEMM: C[M,N] = A[M,K] * B[N,K]^T ----------------
// Block: 256 threads = 8 waves; block tile 256(M) x 64(N); wave tile 32x64.
// B tile (64x32) staged in double-buffered LDS (async path when available).
__global__ __launch_bounds__(256) void gemm_bf16_nt(const unsigned short* __restrict__ A,
                                                    const unsigned short* __restrict__ Bm,
                                                    float* __restrict__ C,
                                                    int M, int N, int K) {
    (void)M;
    __shared__ __align__(16) unsigned short btile[2][64 * 32];
    int tid  = threadIdx.x;
    int lane = tid & 31;
    int wid  = tid >> 5;                         // 0..7
    int m0 = blockIdx.y * 256 + wid * 32;
    int n0 = blockIdx.x * 64;

    v8f acc[2][4] = {};

    stage_b_tile(Bm, btile[0], n0, 0, K, tid);
    int p = 0;
    for (int kk = 0; kk < K; kk += 32) {
        wait_stage();
        __syncthreads();                         // btile[p] ready for all waves
        if (kk + 32 < K) stage_b_tile(Bm, btile[p ^ 1], n0, kk + 32, K, tid);

        v16bf a0 = load_frag_k8(A, K, m0,      kk, lane);
        v16bf a1 = load_frag_k8(A, K, m0 + 16, kk, lane);
        #pragma unroll
        for (int j = 0; j < 4; ++j) {
            v16bf bf = load_frag_lds(btile[p], j * 16, lane);
            acc[0][j] = wmma_bf16(a0, bf, acc[0][j]);
            acc[1][j] = wmma_bf16(a1, bf, acc[1][j]);
        }
        __syncthreads();                         // done reading btile[p]
        p ^= 1;
    }

    int r0 = lane & 15, hi = (lane >> 4) & 1;
    #pragma unroll
    for (int half = 0; half < 2; ++half) {
        #pragma unroll
        for (int r = 0; r < 8; ++r) {
            int row = m0 + half * 16 + r + hi * 8;
            float* crow = C + (size_t)row * N + n0 + r0;
            crow[0]  = acc[half][0][r];
            crow[16] = acc[half][1][r];
            crow[32] = acc[half][2][r];
            crow[48] = acc[half][3][r];
        }
    }
}

// ---------------- RoPE + repack ----------------
// q: [B*T, DIM] fp32 -> [B,H,T,hd] bf16, pre-scaled by 1/sqrt(hd)
__global__ void rope_pack_q(const float* __restrict__ qf, unsigned short* __restrict__ qb) {
    int i = blockIdx.x * 256 + threadIdx.x;
    if (i >= MROWS * DIMC) return;
    int d   = i & (HD - 1);
    int h   = (i >> 6) & (NH - 1);
    int row = i >> 11;                    // b*T + t
    int t   = row & (TT - 1);
    int b   = row >> 11;
    float x0 = qf[i];
    int d2 = (d < 32) ? d + 32 : d - 32;
    float xr = qf[(size_t)row * DIMC + h * HD + d2];
    float rot = (d < 32) ? -xr : xr;
    float inv = __expf(-((float)(2 * (d & 31)) / 64.0f) * 9.2103403719761836f); // ln(10000)
    float ang = (float)t * inv;
    float s, c;
    sincosf(ang, &s, &c);
    float val = (x0 * c + rot * s) * 0.125f;   // 1/sqrt(64)
    qb[(((size_t)b * NH + h) * TT + t) * HD + d] = f2bf(val);
}

// k: [B*T, 512] fp32 -> [B,Hk,T,hd] bf16 (no scale)
__global__ void rope_pack_k(const float* __restrict__ kf, unsigned short* __restrict__ kb) {
    int i = blockIdx.x * 256 + threadIdx.x;
    if (i >= MROWS * KVDIM) return;
    int d   = i & (HD - 1);
    int hk  = (i >> 6) & (NKV - 1);
    int row = i >> 9;                     // b*T + t
    int t   = row & (TT - 1);
    int b   = row >> 11;
    float x0 = kf[i];
    int d2 = (d < 32) ? d + 32 : d - 32;
    float xr = kf[(size_t)row * KVDIM + hk * HD + d2];
    float rot = (d < 32) ? -xr : xr;
    float inv = __expf(-((float)(2 * (d & 31)) / 64.0f) * 9.2103403719761836f);
    float ang = (float)t * inv;
    float s, c;
    sincosf(ang, &s, &c);
    kb[(((size_t)b * NKV + hk) * TT + t) * HD + d] = f2bf(x0 * c + rot * s);
}

// v: [B*T, 512] fp32 -> V^T [B,Hk,hd,T] bf16
__global__ void pack_vt(const float* __restrict__ vf, unsigned short* __restrict__ vtb) {
    int i = blockIdx.x * 256 + threadIdx.x;
    if (i >= MROWS * KVDIM) return;
    int d   = i & (HD - 1);
    int hk  = (i >> 6) & (NKV - 1);
    int row = i >> 9;
    int t   = row & (TT - 1);
    int b   = row >> 11;
    vtb[(((size_t)b * NKV + hk) * HD + d) * TT + t] = f2bf(vf[i]);
}

// ---------------- flash attention (online softmax, causal) ----------------
__global__ __launch_bounds__(128) void flash_attn(const unsigned short* __restrict__ qb,
                                                  const unsigned short* __restrict__ kb,
                                                  const unsigned short* __restrict__ vtb,
                                                  unsigned short* __restrict__ ob) {
    __shared__ __align__(16) unsigned short plds[4][16][32];   // per-wave P tile
    int lane = threadIdx.x & 31;
    int wid  = threadIdx.x >> 5;
    int b  = blockIdx.z;
    int h  = blockIdx.y;
    int hk = h >> 2;                              // rep = NH/NKV = 4
    int q0 = blockIdx.x * 64 + wid * 16;

    const unsigned short* qp = qb  + (((size_t)b * NH  + h ) * TT) * HD;
    const unsigned short* kp = kb  + (((size_t)b * NKV + hk) * TT) * HD;
    const unsigned short* vp = vtb + (((size_t)b * NKV + hk) * HD) * TT;

    v16bf qa0 = load_frag_k8(qp, HD, q0, 0,  lane);   // hd 0..31
    v16bf qa1 = load_frag_k8(qp, HD, q0, 32, lane);   // hd 32..63

    v8f o0 = {}, o1 = {}, o2 = {}, o3 = {};
    float mrow[8], lrow[8];
    #pragma unroll
    for (int r = 0; r < 8; ++r) { mrow[r] = NEGBIG; lrow[r] = 0.f; }

    int r0 = lane & 15, hi = (lane >> 4) & 1;

    for (int kt = 0; kt < q0 + 16; kt += 32) {
        // ---- scores: S[16q x 32k] as two 16x16 D tiles ----
        v16bf k0l = load_frag_k8(kp, HD, kt,      0,  lane);
        v16bf k0h = load_frag_k8(kp, HD, kt,      32, lane);
        v16bf k1l = load_frag_k8(kp, HD, kt + 16, 0,  lane);
        v16bf k1h = load_frag_k8(kp, HD, kt + 16, 32, lane);
        v8f s0 = {}, s1 = {};
        s0 = wmma_bf16(qa0, k0l, s0);
        s0 = wmma_bf16(qa1, k0h, s0);
        s1 = wmma_bf16(qa0, k1l, s1);
        s1 = wmma_bf16(qa1, k1h, s1);

        int key0 = kt + r0, key1 = key0 + 16;
        float nm[8];
        #pragma unroll
        for (int r = 0; r < 8; ++r) {
            int qq = q0 + r + hi * 8;
            float a = (key0 <= qq) ? s0[r] : NEGBIG;
            float c = (key1 <= qq) ? s1[r] : NEGBIG;
            s0[r] = a; s1[r] = c;
            float mx = fmaxf(a, c);
            mx = fmaxf(mx, __shfl_xor(mx, 1));
            mx = fmaxf(mx, __shfl_xor(mx, 2));
            mx = fmaxf(mx, __shfl_xor(mx, 4));
            mx = fmaxf(mx, __shfl_xor(mx, 8));    // stays within 16-lane half
            nm[r] = fmaxf(mrow[r], mx);
        }
        #pragma unroll
        for (int r = 0; r < 8; ++r) {
            float p0 = __expf(s0[r] - nm[r]);
            float p1 = __expf(s1[r] - nm[r]);
            float ps = p0 + p1;
            ps += __shfl_xor(ps, 1);
            ps += __shfl_xor(ps, 2);
            ps += __shfl_xor(ps, 4);
            ps += __shfl_xor(ps, 8);
            float alpha = __expf(mrow[r] - nm[r]);
            lrow[r] = lrow[r] * alpha + ps;
            mrow[r] = nm[r];
            o0[r] *= alpha; o1[r] *= alpha; o2[r] *= alpha; o3[r] *= alpha;
            int rw = r + hi * 8;
            plds[wid][rw][r0]      = f2bf(p0);
            plds[wid][rw][r0 + 16] = f2bf(p1);
        }
        __builtin_amdgcn_wave_barrier();   // LDS ops are in-order per wave; stop reordering

        // ---- P (A frag from LDS) x V^T (B frags) ----
        FragBF pf;
        const unsigned short* pr = &plds[wid][r0][hi * 8];
        pf.u2[0] = *reinterpret_cast<const uint4*>(pr);
        pf.u2[1] = *reinterpret_cast<const uint4*>(pr + 16);
        v16bf pa = pf.v;
        v16bf vb0 = load_frag_k8(vp, TT, 0,  kt, lane);
        v16bf vb1 = load_frag_k8(vp, TT, 16, kt, lane);
        v16bf vb2 = load_frag_k8(vp, TT, 32, kt, lane);
        v16bf vb3 = load_frag_k8(vp, TT, 48, kt, lane);
        o0 = wmma_bf16(pa, vb0, o0);
        o1 = wmma_bf16(pa, vb1, o1);
        o2 = wmma_bf16(pa, vb2, o2);
        o3 = wmma_bf16(pa, vb3, o3);
    }

    #pragma unroll
    for (int r = 0; r < 8; ++r) {
        float inv = 1.0f / lrow[r];
        int row = q0 + r + hi * 8;
        size_t base = ((size_t)b * TT + row) * DIMC + h * HD + r0;
        ob[base]      = f2bf(o0[r] * inv);
        ob[base + 16] = f2bf(o1[r] * inv);
        ob[base + 32] = f2bf(o2[r] * inv);
        ob[base + 48] = f2bf(o3[r] * inv);
    }
}

extern "C" void kernel_launch(void* const* d_in, const int* in_sizes, int n_in,
                              void* d_out, int out_size, void* d_ws, size_t ws_size,
                              hipStream_t stream) {
    (void)in_sizes; (void)n_in; (void)out_size; (void)ws_size;
    const float* x  = (const float*)d_in[0];
    const float* wq = (const float*)d_in[1];
    const float* wk = (const float*)d_in[2];
    const float* wv = (const float*)d_in[3];
    const float* wo = (const float*)d_in[4];
    float* out = (float*)d_out;

    char* ws = (char*)d_ws;
    size_t off = 0;
    auto alloc = [&](size_t bytes) -> char* {
        char* p = ws + off;
        off = (off + bytes + 255) & ~(size_t)255;
        return p;
    };
    unsigned short* xb  = (unsigned short*)alloc((size_t)MROWS * DIMC  * 2);
    unsigned short* wqb = (unsigned short*)alloc((size_t)DIMC  * DIMC  * 2);
    unsigned short* wkb = (unsigned short*)alloc((size_t)KVDIM * DIMC  * 2);
    unsigned short* wvb = (unsigned short*)alloc((size_t)KVDIM * DIMC  * 2);
    unsigned short* wob = (unsigned short*)alloc((size_t)DIMC  * DIMC  * 2);
    float*          qf  = (float*)         alloc((size_t)MROWS * DIMC  * 4);
    float*          kf  = (float*)         alloc((size_t)MROWS * KVDIM * 4);
    float*          vf  = (float*)         alloc((size_t)MROWS * KVDIM * 4);
    unsigned short* qb  = (unsigned short*)alloc((size_t)MROWS * DIMC  * 2);
    unsigned short* kbp = (unsigned short*)alloc((size_t)MROWS * KVDIM * 2);
    unsigned short* vtb = (unsigned short*)alloc((size_t)MROWS * KVDIM * 2);
    unsigned short* ob  = (unsigned short*)alloc((size_t)MROWS * DIMC  * 2);

    // 1) fp32 -> bf16 conversions
    int nx = MROWS * DIMC, nw = DIMC * DIMC, nkv = KVDIM * DIMC;
    cvt_f32_bf16<<<(nx  + 255) / 256, 256, 0, stream>>>(x,  xb,  nx);
    cvt_f32_bf16<<<(nw  + 255) / 256, 256, 0, stream>>>(wq, wqb, nw);
    cvt_f32_bf16<<<(nkv + 255) / 256, 256, 0, stream>>>(wk, wkb, nkv);
    cvt_f32_bf16<<<(nkv + 255) / 256, 256, 0, stream>>>(wv, wvb, nkv);
    cvt_f32_bf16<<<(nw  + 255) / 256, 256, 0, stream>>>(wo, wob, nw);

    // 2) QKV projections (NT GEMMs, fp32 accumulate)
    gemm_bf16_nt<<<dim3(DIMC / 64,  MROWS / 256), 256, 0, stream>>>(xb, wqb, qf, MROWS, DIMC,  DIMC);
    gemm_bf16_nt<<<dim3(KVDIM / 64, MROWS / 256), 256, 0, stream>>>(xb, wkb, kf, MROWS, KVDIM, DIMC);
    gemm_bf16_nt<<<dim3(KVDIM / 64, MROWS / 256), 256, 0, stream>>>(xb, wvb, vf, MROWS, KVDIM, DIMC);

    // 3) RoPE + repack to attention layouts
    rope_pack_q<<<(MROWS * DIMC  + 255) / 256, 256, 0, stream>>>(qf, qb);
    rope_pack_k<<<(MROWS * KVDIM + 255) / 256, 256, 0, stream>>>(kf, kbp);
    pack_vt    <<<(MROWS * KVDIM + 255) / 256, 256, 0, stream>>>(vf, vtb);

    // 4) causal flash attention
    flash_attn<<<dim3(TT / 64, NH, BB), 128, 0, stream>>>(qb, kbp, vtb, ob);

    // 5) output projection
    gemm_bf16_nt<<<dim3(DIMC / 64, MROWS / 256), 256, 0, stream>>>(ob, wob, out, MROWS, DIMC, DIMC);
}
